// GraphNetwork_75239237091740
// MI455X (gfx1250) — matrix operand order repaired and checked by reading, not compiled, not run on hardware
//
#include <hip/hip_runtime.h>
#include <hip/hip_bf16.h>

typedef __attribute__((ext_vector_type(16))) _Float16 v16h;
typedef __attribute__((ext_vector_type(8)))  _Float16 v8h;
typedef __attribute__((ext_vector_type(4)))  _Float16 v4h;
typedef __attribute__((ext_vector_type(8)))  float    v8f;
typedef __attribute__((ext_vector_type(4)))  float    v4f;

#define NNODES 50000
#define NEDGES 600000
#define DD 128
#define HH 256

// ---------------------------------------------------------------------------
// Helpers
// ---------------------------------------------------------------------------
static __device__ __forceinline__ v16h cat8(v8h lo, v8h hi) {
  return __builtin_shufflevector(lo, hi, 0,1,2,3,4,5,6,7,8,9,10,11,12,13,14,15);
}

// A-fragment (16x32 f16, M = lane&15) from an LDS row of f16.
// ISA 7.12.2: lanes 0-15: halves 0..7 -> K=kb+0..7, halves 8..15 -> K=kb+16..23
//             lanes 16-31: shifted by +8.
static __device__ __forceinline__ v16h load_afrag(const _Float16* rowp, int kb, int lane) {
  int k0 = kb + ((lane >= 16) ? 8 : 0);
  v8h lo = *(const v8h*)(rowp + k0);
  v8h hi = *(const v8h*)(rowp + k0 + 16);
  return cat8(lo, hi);
}

// B-fragment (32x16 f16) from pre-packed weight tiles: tile (kt,nt) stores
// 512 halves, lane-major, 16 contiguous halves (32B) per lane.
static __device__ __forceinline__ v16h load_bfrag(const _Float16* __restrict__ wt,
                                                  int numKT, int kt, int nt, int lane) {
  const _Float16* p = wt + (((size_t)(nt * numKT + kt)) << 9) + (lane << 4);
  return *(const v16h*)p;
}

static __device__ __forceinline__ v8f wmma_f16(v16h a, v16h b, v8f c) {
  return __builtin_amdgcn_wmma_f32_16x16x32_f16(false, a, false, b, (short)0, c, false, false);
}

// ---------------------------------------------------------------------------
// Kernel: zero the aggregate buffers (float4 granularity)
// ---------------------------------------------------------------------------
__global__ __launch_bounds__(256) void zero_k(float* __restrict__ p, long n4) {
  long i = (long)blockIdx.x * blockDim.x + threadIdx.x;
  if (i < n4) {
    v4f z = {0.f, 0.f, 0.f, 0.f};
    *(v4f*)(p + i * 4) = z;
  }
}

// ---------------------------------------------------------------------------
// Kernel: pack f32 weight [K x N] into f16 WMMA B-fragment tiles.
// Element (lane,h) of tile(kt,nt): k = kt*32 + (lane>=16?8:0) + (h<8 ? h : h+8), n = nt*16 + (lane&15)
// ---------------------------------------------------------------------------
__global__ __launch_bounds__(256) void pack_w(const float* __restrict__ W,
                                              _Float16* __restrict__ out,
                                              int K, int N) {
  int gid = blockIdx.x * blockDim.x + threadIdx.x;
  if (gid >= K * N) return;
  int tile   = gid >> 9;
  int within = gid & 511;
  int lane   = within >> 4;
  int h      = within & 15;
  int numKT  = K >> 5;
  int kt     = tile % numKT;
  int nt     = tile / numKT;
  int k = kt * 32 + ((lane >= 16) ? 8 : 0) + (h < 8 ? h : h + 8);
  int n = nt * 16 + (lane & 15);
  out[gid] = (_Float16)W[(size_t)k * N + n];
}

// ---------------------------------------------------------------------------
// Shared compute core: given xs[64x384] staged in LDS, compute
// hs = relu(xs @ W1 + b1) (64x256), then acc2 = hs @ W2 + b2 (64x128).
// Layer 1 fuses both n-tiles of this wave into one k-loop: each A fragment
// fetched from LDS once feeds 2 WMMAs; each B fragment feeds 4 WMMAs.
// ---------------------------------------------------------------------------
struct L2Out { v8f acc[4]; int col; };

static __device__ __forceinline__ L2Out mlp_core(
    const _Float16* __restrict__ xs,   // LDS, 64x384
    _Float16* __restrict__ hs,         // LDS, 64x256
    const _Float16* __restrict__ w1t, const float* __restrict__ b1,
    const _Float16* __restrict__ w2t, const float* __restrict__ b2,
    int wave, int lane)
{
  // ---- layer 1: [64x384] @ [384x256]; this wave: nt0 = wave, nt1 = wave+8 ----
  {
    const int nt0 = wave;
    const int nt1 = wave + 8;
    const float bias0 = b1[nt0 * 16 + (lane & 15)];
    const float bias1 = b1[nt1 * 16 + (lane & 15)];
    v8f acc0[4], acc1[4];
    #pragma unroll
    for (int g = 0; g < 4; ++g)
      #pragma unroll
      for (int r = 0; r < 8; ++r) { acc0[g][r] = bias0; acc1[g][r] = bias1; }

    const _Float16* rowp = &xs[(lane & 15) * 384];
    #pragma unroll
    for (int kt = 0; kt < 12; ++kt) {
      v16h b0 = load_bfrag(w1t, 12, kt, nt0, lane);
      v16h b1v = load_bfrag(w1t, 12, kt, nt1, lane);
      // all 4 A fragments up-front in distinct registers -> ds loads pipeline
      v16h a0 = load_afrag(rowp + 0 * (16 * 384), kt * 32, lane);
      v16h a1 = load_afrag(rowp + 1 * (16 * 384), kt * 32, lane);
      v16h a2 = load_afrag(rowp + 2 * (16 * 384), kt * 32, lane);
      v16h a3 = load_afrag(rowp + 3 * (16 * 384), kt * 32, lane);
      acc0[0] = wmma_f16(a0, b0, acc0[0]);
      acc0[1] = wmma_f16(a1, b0, acc0[1]);
      acc0[2] = wmma_f16(a2, b0, acc0[2]);
      acc0[3] = wmma_f16(a3, b0, acc0[3]);
      acc1[0] = wmma_f16(a0, b1v, acc1[0]);
      acc1[1] = wmma_f16(a1, b1v, acc1[1]);
      acc1[2] = wmma_f16(a2, b1v, acc1[2]);
      acc1[3] = wmma_f16(a3, b1v, acc1[3]);
    }
    const int col0 = nt0 * 16 + (lane & 15);
    const int col1 = nt1 * 16 + (lane & 15);
    #pragma unroll
    for (int g = 0; g < 4; ++g) {
      int rowb = g * 16 + 8 * (lane >> 4);
      #pragma unroll
      for (int r = 0; r < 8; ++r) {
        float v0 = acc0[g][r]; v0 = v0 > 0.f ? v0 : 0.f;   // ReLU
        float v1 = acc1[g][r]; v1 = v1 > 0.f ? v1 : 0.f;
        hs[(rowb + r) * 256 + col0] = (_Float16)v0;
        hs[(rowb + r) * 256 + col1] = (_Float16)v1;
      }
    }
  }
  __syncthreads();

  // ---- layer 2: [64x256] @ [256x128]; this wave: nt = wave ----
  L2Out o;
  {
    const int nt = wave;
    const float bias = b2[nt * 16 + (lane & 15)];
    #pragma unroll
    for (int g = 0; g < 4; ++g)
      #pragma unroll
      for (int r = 0; r < 8; ++r) o.acc[g][r] = bias;
    const _Float16* rowp = &hs[(lane & 15) * 256];
    #pragma unroll
    for (int kt = 0; kt < 8; ++kt) {
      v16h b = load_bfrag(w2t, 8, kt, nt, lane);
      v16h a0 = load_afrag(rowp + 0 * (16 * 256), kt * 32, lane);
      v16h a1 = load_afrag(rowp + 1 * (16 * 256), kt * 32, lane);
      v16h a2 = load_afrag(rowp + 2 * (16 * 256), kt * 32, lane);
      v16h a3 = load_afrag(rowp + 3 * (16 * 256), kt * 32, lane);
      o.acc[0] = wmma_f16(a0, b, o.acc[0]);
      o.acc[1] = wmma_f16(a1, b, o.acc[1]);
      o.acc[2] = wmma_f16(a2, b, o.acc[2]);
      o.acc[3] = wmma_f16(a3, b, o.acc[3]);
    }
    o.col = nt * 16 + (lane & 15);
  }
  return o;
}

// ---------------------------------------------------------------------------
// Edge kernel: 64 edges / block, 8 waves.
// ---------------------------------------------------------------------------
__global__ __launch_bounds__(256) void edge_k(
    const float* __restrict__ node_feat,
    const float* __restrict__ edge_feat,
    const int*   __restrict__ senders,
    const int*   __restrict__ receivers,
    const _Float16* __restrict__ w1t, const float* __restrict__ b1,
    const _Float16* __restrict__ w2t, const float* __restrict__ b2,
    float* __restrict__ out_edges,
    float* __restrict__ sent_agg,
    float* __restrict__ recv_agg)
{
  __shared__ _Float16 xs[64 * 384];   // 48 KB input activations (f16)
  __shared__ _Float16 hs[64 * 256];   // 32 KB hidden activations (f16)
  __shared__ int sIdx[64];
  __shared__ int rIdx[64];

  const int tid   = threadIdx.x;
  const int lane  = tid & 31;
  const int wave  = tid >> 5;
  const int eBase = blockIdx.x * 64;

  if (tid < 64) {
    sIdx[tid] = senders[eBase + tid];
    rIdx[tid] = receivers[eBase + tid];
  }
  __syncthreads();

  // ---- gather + f32->f16 convert: 64 rows x 96 float4 chunks ----
  #pragma unroll
  for (int i = 0; i < 24; ++i) {
    int c      = tid + 256 * i;       // 0..6143
    int row    = c / 96;
    int within = c % 96;              // float4 index within row
    int seg    = within >> 5;         // 0: edge, 1: sender, 2: receiver
    int off    = (within & 31) << 2;  // float offset within 128
    v4f v;
    if (seg == 0) {
      // edge_feat streams through exactly once: non-temporal to keep
      // node_feat + weights resident in L2.
      v = __builtin_nontemporal_load(
            (const v4f*)(edge_feat + (size_t)(eBase + row) * DD + off));
    } else if (seg == 1) {
      v = *(const v4f*)(node_feat + (size_t)sIdx[row] * DD + off);
    } else {
      v = *(const v4f*)(node_feat + (size_t)rIdx[row] * DD + off);
    }
    v4h hv = {(_Float16)v[0], (_Float16)v[1], (_Float16)v[2], (_Float16)v[3]};
    *(v4h*)&xs[row * 384 + within * 4] = hv;
  }
  __syncthreads();

  L2Out o = mlp_core(xs, hs, w1t, b1, w2t, b2, wave, lane);

  #pragma unroll
  for (int g = 0; g < 4; ++g) {
    int rowb = g * 16 + 8 * (lane >> 4);
    #pragma unroll
    for (int r = 0; r < 8; ++r) {
      int row = rowb + r;
      int e   = eBase + row;
      float v = o.acc[g][r];
      __builtin_nontemporal_store(v, &out_edges[(size_t)e * DD + o.col]);
      atomicAdd(&sent_agg[(size_t)sIdx[row] * DD + o.col], v);
      atomicAdd(&recv_agg[(size_t)rIdx[row] * DD + o.col], v);
    }
  }
}

// ---------------------------------------------------------------------------
// Node kernel: 64 nodes / block.
// ---------------------------------------------------------------------------
__global__ __launch_bounds__(256) void node_k(
    const float* __restrict__ node_feat,
    const float* __restrict__ sent_agg,
    const float* __restrict__ recv_agg,
    const _Float16* __restrict__ w1t, const float* __restrict__ b1,
    const _Float16* __restrict__ w2t, const float* __restrict__ b2,
    float* __restrict__ out_nodes)
{
  __shared__ _Float16 xs[64 * 384];
  __shared__ _Float16 hs[64 * 256];

  const int tid   = threadIdx.x;
  const int lane  = tid & 31;
  const int wave  = tid >> 5;
  const int nBase = blockIdx.x * 64;

  #pragma unroll
  for (int i = 0; i < 24; ++i) {
    int c      = tid + 256 * i;
    int row    = c / 96;
    int within = c % 96;
    int seg    = within >> 5;
    int off    = (within & 31) << 2;
    int n      = nBase + row;
    v4f v = {0.f, 0.f, 0.f, 0.f};
    if (n < NNODES) {
      const float* src = (seg == 0 ? node_feat : (seg == 1 ? sent_agg : recv_agg))
                         + (size_t)n * DD + off;
      v = *(const v4f*)src;
    }
    v4h hv = {(_Float16)v[0], (_Float16)v[1], (_Float16)v[2], (_Float16)v[3]};
    *(v4h*)&xs[row * 384 + within * 4] = hv;
  }
  __syncthreads();

  L2Out o = mlp_core(xs, hs, w1t, b1, w2t, b2, wave, lane);

  #pragma unroll
  for (int g = 0; g < 4; ++g) {
    int rowb = g * 16 + 8 * (lane >> 4);
    #pragma unroll
    for (int r = 0; r < 8; ++r) {
      int n = nBase + rowb + r;
      if (n < NNODES)
        __builtin_nontemporal_store(o.acc[g][r], &out_nodes[(size_t)n * DD + o.col]);
    }
  }
}

// ---------------------------------------------------------------------------
// Launch
// ---------------------------------------------------------------------------
extern "C" void kernel_launch(void* const* d_in, const int* in_sizes, int n_in,
                              void* d_out, int out_size, void* d_ws, size_t ws_size,
                              hipStream_t stream) {
  const float* node_feat = (const float*)d_in[0];
  const float* edge_feat = (const float*)d_in[1];
  const int*   senders   = (const int*)  d_in[2];
  const int*   receivers = (const int*)  d_in[3];
  const float* eW1 = (const float*)d_in[4];
  const float* eb1 = (const float*)d_in[5];
  const float* eW2 = (const float*)d_in[6];
  const float* eb2 = (const float*)d_in[7];
  const float* nW1 = (const float*)d_in[8];
  const float* nb1 = (const float*)d_in[9];
  const float* nW2 = (const float*)d_in[10];
  const float* nb2 = (const float*)d_in[11];

  // Workspace layout (needs ~52 MB):
  //  [0, 192K)        eW1 f16 tiles   (384*256*2)
  //  [192K, 256K)     eW2 f16 tiles   (256*128*2)
  //  [256K, 448K)     nW1 f16 tiles
  //  [448K, 512K)     nW2 f16 tiles
  //  [512K, ...)      sent_agg f32 [N,D] ; recv_agg f32 [N,D]
  char* ws = (char*)d_ws;
  _Float16* eW1t = (_Float16*)(ws + 0);
  _Float16* eW2t = (_Float16*)(ws + 196608);
  _Float16* nW1t = (_Float16*)(ws + 262144);
  _Float16* nW2t = (_Float16*)(ws + 458752);
  float* sent_agg = (float*)(ws + 524288);
  float* recv_agg = sent_agg + (size_t)NNODES * DD;

  float* out_nodes = (float*)d_out;
  float* out_edges = out_nodes + (size_t)NNODES * DD;

  // 1) zero aggregates (contiguous 2*N*D floats)
  long n4 = (long)2 * NNODES * DD / 4;
  zero_k<<<(int)((n4 + 255) / 256), 256, 0, stream>>>(sent_agg, n4);

  // 2) pack weights to f16 WMMA fragments
  pack_w<<<(384 * 256 + 255) / 256, 256, 0, stream>>>(eW1, eW1t, 384, 256);
  pack_w<<<(256 * 128 + 255) / 256, 256, 0, stream>>>(eW2, eW2t, 256, 128);
  pack_w<<<(384 * 256 + 255) / 256, 256, 0, stream>>>(nW1, nW1t, 384, 256);
  pack_w<<<(256 * 128 + 255) / 256, 256, 0, stream>>>(nW2, nW2t, 256, 128);

  // 3) edge MLP + aggregation (600000 / 64 = 9375 blocks)
  edge_k<<<NEDGES / 64, 256, 0, stream>>>(node_feat, edge_feat, senders, receivers,
                                          eW1t, eb1, eW2t, eb2,
                                          out_edges, sent_agg, recv_agg);

  // 4) node MLP (ceil(50000/64) = 782 blocks)
  node_k<<<(NNODES + 63) / 64, 256, 0, stream>>>(node_feat, sent_agg, recv_agg,
                                                 nW1t, nb1, nW2t, nb2,
                                                 out_nodes);
}
